// BertCLModel_44513041056396
// MI455X (gfx1250) — matrix operand order repaired and checked by reading, not compiled, not run on hardware
//
#include <hip/hip_runtime.h>
#include <hip/hip_bf16.h>
#include <math.h>

// ---------------------------------------------------------------------------
// Problem constants (from reference): B=256, S=512, H=768, TAU=0.5, n=B/4=64
// ---------------------------------------------------------------------------
#define BATCH 256
#define SEQ   512
#define HID   768
#define NCH   8      // sequence split for pooling partials
#define SCH   (SEQ / NCH)   // 64
#define NROW  64     // n = B//4 ; only logits rows [0,64) are needed

typedef __attribute__((ext_vector_type(16))) __bf16 v16bf;
typedef __attribute__((ext_vector_type(8)))  float  v8f;

union V16 { v16bf v; uint4 q[2]; };

__device__ __forceinline__ unsigned short f32_to_bf16_rne(float f) {
    unsigned u = __float_as_uint(f);
    unsigned r = u + 0x7FFFu + ((u >> 16) & 1u);   // round-to-nearest-even
    return (unsigned short)(r >> 16);
}

// ---------------------------------------------------------------------------
// Kernel 1: bandwidth-bound pooling. grid = B*NCH blocks, 192 threads.
// Each block sums SCH=64 rows of H=768 floats via float4 loads -> partial.
// 402.7 MB streamed once; 2048 blocks keep HBM saturated.
// ---------------------------------------------------------------------------
__global__ void __launch_bounds__(192)
k_pool_partial(const float* __restrict__ x, float* __restrict__ partial) {
    int b = blockIdx.x >> 3;
    int c = blockIdx.x & 7;
    int t = threadIdx.x;                 // 0..191, one float4 (4 hidden dims)
    const float4* base = (const float4*)(x + ((size_t)b * SEQ + (size_t)c * SCH) * HID);
    float4 a = make_float4(0.f, 0.f, 0.f, 0.f);
#pragma unroll 4
    for (int s = 0; s < SCH; ++s) {
        float4 v = base[(size_t)s * (HID / 4) + t];
        a.x += v.x; a.y += v.y; a.z += v.z; a.w += v.w;
    }
    float4* p = (float4*)(partial + ((size_t)b * NCH + c) * HID);
    p[t] = a;
}

// ---------------------------------------------------------------------------
// Kernel 2: per-row (mask-sum, pooled, L2-norm) then split-bf16 encode.
// grid = B blocks, 256 threads (each owns 3 hidden dims).
// ---------------------------------------------------------------------------
__global__ void __launch_bounds__(256)
k_norm_split(const float* __restrict__ partial, const float* __restrict__ mask,
             unsigned short* __restrict__ zhi, unsigned short* __restrict__ zlo) {
    __shared__ float red[256];
    int b = blockIdx.x, t = threadIdx.x;

    // mask row sum (faithful to reference: divide by mask.sum)
    float ml = mask[b * SEQ + t] + mask[b * SEQ + 256 + t];
    red[t] = ml; __syncthreads();
    for (int off = 128; off > 0; off >>= 1) {
        if (t < off) red[t] += red[t + off];
        __syncthreads();
    }
    float msum = red[0];
    __syncthreads();

    // pooled = sum over the NCH partials, divided by msum; local sum of squares
    float p[3]; float ss = 0.f;
#pragma unroll
    for (int u = 0; u < 3; ++u) {
        int h = t + u * 256;
        float s = 0.f;
#pragma unroll
        for (int c = 0; c < NCH; ++c) s += partial[((size_t)b * NCH + c) * HID + h];
        p[u] = s / msum;
        ss += p[u] * p[u];
    }
    red[t] = ss; __syncthreads();
    for (int off = 128; off > 0; off >>= 1) {
        if (t < off) red[t] += red[t + off];
        __syncthreads();
    }
    float rn = 1.0f / sqrtf(red[0]);

#pragma unroll
    for (int u = 0; u < 3; ++u) {
        int h = t + u * 256;
        float z = p[u] * rn;
        unsigned short hb = f32_to_bf16_rne(z);
        float hf = __uint_as_float(((unsigned)hb) << 16);
        unsigned short lb = f32_to_bf16_rne(z - hf);
        zhi[(size_t)b * HID + h] = hb;
        zlo[(size_t)b * HID + h] = lb;
    }
}

// ---------------------------------------------------------------------------
// Kernel 3: Gram slab logits[0:64, 0:256] = (z z^T)/tau via split-bf16 WMMA.
// One wave per 16x16 tile; 64 tiles; K-loop 768/32 = 24 steps, 3 WMMAs each.
// acc_f32 += hi*hi^T + hi*lo^T + lo*hi^T  (~fp32-accurate).
// ---------------------------------------------------------------------------
__global__ void __launch_bounds__(32)
k_gram_wmma(const unsigned short* __restrict__ zhi,
            const unsigned short* __restrict__ zlo,
            float* __restrict__ logits) {
    int tile = blockIdx.x;          // 0..63
    int ti = tile >> 4;             // output row tile 0..3  (rows 0..63)
    int tj = tile & 15;             // output col tile 0..15 (cols 0..255)
    int lane = threadIdx.x & 31;
    int half = lane >> 4;
    int idx  = lane & 15;

    const unsigned short* ahiP = zhi + (size_t)(ti * 16 + idx) * HID;
    const unsigned short* aloP = zlo + (size_t)(ti * 16 + idx) * HID;
    const unsigned short* bhiP = zhi + (size_t)(tj * 16 + idx) * HID;
    const unsigned short* bloP = zlo + (size_t)(tj * 16 + idx) * HID;

    v8f acc = {};
    for (int k = 0; k < HID; k += 32) {
        V16 ah, al, bh, bl;
        // A 16x32 bf16: lane(half h) holds K in [8h,8h+8) and [16+8h,16+8h+8)
        ah.q[0] = *(const uint4*)(ahiP + k + 8 * half);
        ah.q[1] = *(const uint4*)(ahiP + k + 16 + 8 * half);
        al.q[0] = *(const uint4*)(aloP + k + 8 * half);
        al.q[1] = *(const uint4*)(aloP + k + 16 + 8 * half);
        // B 32x16 bf16: lane(half h) holds K in [8h,8h+8) and [16+8h,16+8h+8)
        bh.q[0] = *(const uint4*)(bhiP + k + 8 * half);
        bh.q[1] = *(const uint4*)(bhiP + k + 16 + 8 * half);
        bl.q[0] = *(const uint4*)(bloP + k + 8 * half);
        bl.q[1] = *(const uint4*)(bloP + k + 16 + 8 * half);

        acc = __builtin_amdgcn_wmma_f32_16x16x32_bf16(false, ah.v, false, bh.v,
                                                      (short)0, acc, false, false);
        acc = __builtin_amdgcn_wmma_f32_16x16x32_bf16(false, ah.v, false, bl.v,
                                                      (short)0, acc, false, false);
        acc = __builtin_amdgcn_wmma_f32_16x16x32_bf16(false, al.v, false, bh.v,
                                                      (short)0, acc, false, false);
    }

    // D layout: VGPR r -> row = r + 8*half ; lane idx -> col. Scale by 1/TAU=2.
#pragma unroll
    for (int r = 0; r < 8; ++r) {
        int row = ti * 16 + r + 8 * half;
        int col = tj * 16 + idx;
        logits[row * BATCH + col] = acc[r] * 2.0f;
    }
}

// ---------------------------------------------------------------------------
// Kernel 4: per row i<64: logden = logsumexp_{k!=i}, then
// rowres[i] = sum_{i<j<64} (logden - logits[i][j]).  grid=64, block=256.
// ---------------------------------------------------------------------------
__global__ void __launch_bounds__(256)
k_row_reduce(const float* __restrict__ logits, float* __restrict__ rowres) {
    __shared__ float red[256];
    int i = blockIdx.x, t = threadIdx.x;
    float v  = logits[i * BATCH + t];
    float mv = (t == i) ? -3.4e38f : v;      // mask diagonal

    red[t] = mv; __syncthreads();
    for (int off = 128; off > 0; off >>= 1) {
        if (t < off) red[t] = fmaxf(red[t], red[t + off]);
        __syncthreads();
    }
    float mx = red[0]; __syncthreads();

    red[t] = expf(mv - mx); __syncthreads();
    for (int off = 128; off > 0; off >>= 1) {
        if (t < off) red[t] += red[t + off];
        __syncthreads();
    }
    float logden = mx + logf(red[0]); __syncthreads();

    float contrib = (t > i && t < NROW) ? (logden - v) : 0.0f;
    red[t] = contrib; __syncthreads();
    for (int off = 128; off > 0; off >>= 1) {
        if (t < off) red[t] += red[t + off];
        __syncthreads();
    }
    if (t == 0) rowres[i] = red[0];
}

// ---------------------------------------------------------------------------
// Kernel 5: final scalar: out = -2/n * (n-1) * sum(rowres), n = 64.
// ---------------------------------------------------------------------------
__global__ void __launch_bounds__(64)
k_finalize(const float* __restrict__ rowres, float* __restrict__ out) {
    __shared__ float red[64];
    int t = threadIdx.x;
    red[t] = rowres[t]; __syncthreads();
    for (int off = 32; off > 0; off >>= 1) {
        if (t < off) red[t] += red[t + off];
        __syncthreads();
    }
    if (t == 0) out[0] = (-2.0f / (float)NROW) * (float)(NROW - 1) * red[0];
}

// ---------------------------------------------------------------------------
// Launch
// ---------------------------------------------------------------------------
extern "C" void kernel_launch(void* const* d_in, const int* in_sizes, int n_in,
                              void* d_out, int out_size, void* d_ws, size_t ws_size,
                              hipStream_t stream) {
    const float* x    = (const float*)d_in[0];   // [256, 512, 768] f32
    const float* mask = (const float*)d_in[1];   // [256, 512] f32
    float* out = (float*)d_out;                  // scalar f32

    // Workspace layout (all 16B-aligned):
    float* partial        = (float*)d_ws;                                 // 256*8*768 f32
    unsigned short* zhi   = (unsigned short*)(partial + BATCH * NCH * HID);
    unsigned short* zlo   = zhi + (size_t)BATCH * HID;
    float* logits         = (float*)(zlo + (size_t)BATCH * HID);          // 64*256 f32
    float* rowres         = logits + NROW * BATCH;                        // 64 f32

    k_pool_partial<<<BATCH * NCH, 192, 0, stream>>>(x, partial);
    k_norm_split <<<BATCH, 256, 0, stream>>>(partial, mask, zhi, zlo);
    k_gram_wmma  <<<NROW, 32, 0, stream>>>(zhi, zlo, logits);
    k_row_reduce <<<NROW, 256, 0, stream>>>(logits, rowres);
    k_finalize   <<<1, 64, 0, stream>>>(rowres, out);
}
